// BidirRecurrentModel_64433099374815
// MI455X (gfx1250) — compile-verified, matrix operand order checked by reading, not run on hardware
//
#include <hip/hip_runtime.h>
#include <hip/hip_bf16.h>
#include <stdint.h>

// ---------------------------------------------------------------------------
// Persistent bf16-WMMA LSTM for MI455X (gfx1250).
//   - 16 persistent workgroups, each owns 32 h-columns (=> 128 gate columns,
//     i/f/g/o pre-permuted contiguously at weight-pack time).
//   - Weights packed to bf16, K-major (x-rows then h-rows), L2-resident (8MB).
//   - Per step: TDM DMAs the 64x512 activation tiles into LDS, waves run
//     v_wmma_f32_16x16x32_bf16, gates spilled to LDS, fp32 cell update,
//     h written to ping-pong global buffers, grid sync (cluster barrier if
//     clustered, else L2 atomic barrier).
// ---------------------------------------------------------------------------

#define BB   64
#define TT   1024
#define DD   512
#define HH   512
#define GG   2048     // 4*H
#define KT   1024     // D + H (also 2H for the final FC)
#define OO   512
#define NWG  16       // persistent workgroups
#define NC   32       // h columns owned per WG (NWG*NC == HH)

typedef __bf16 bf16;
typedef unsigned int u32;
typedef unsigned long long u64;
typedef __attribute__((ext_vector_type(8)))  bf16  v8bf;
typedef __attribute__((ext_vector_type(16))) bf16  v16bf;
typedef __attribute__((ext_vector_type(8)))  float v8f;
typedef unsigned int u32x4 __attribute__((ext_vector_type(4)));
typedef int          i32x8 __attribute__((ext_vector_type(8)));
typedef int          i32x4 __attribute__((ext_vector_type(4)));

#if defined(__has_builtin)
#  if __has_builtin(__builtin_amdgcn_tensor_load_to_lds)
#    define USE_TDM 1
#  else
#    define USE_TDM 0
#  endif
#else
#  define USE_TDM 0
#endif

// ---------------------------------------------------------------------------
// WMMA fragment load: 16 bf16 per lane following the ISA 16-bit A/B layout:
// lanes 0-15 hold K = {k..k+7, k+16..k+23}, lanes 16-31 hold +8. Caller folds
// the per-lane +klo; we load the two 8-half chunks at +0 and +16.
// ---------------------------------------------------------------------------
__device__ __forceinline__ v16bf frag_ld(const bf16* p) {
  v8bf lo = *(const v8bf*)p;
  v8bf hi = *(const v8bf*)(p + 16);
  return __builtin_shufflevector(lo, hi, 0,1,2,3,4,5,6,7,8,9,10,11,12,13,14,15);
}

__device__ __forceinline__ u32 lds_off(const void* p) {
  // addrspace(3) offsets live in the low 32 bits of the flat LDS-aperture addr
  return (u32)(u64)p;
}

#if USE_TDM
// Build a 2D Tensor-DMA descriptor (D# group0/group1) and issue
// tensor_load_to_lds: tile = rows x rowElems bf16, global row stride in elems.
// (6-arg builtin on this toolchain: g0, g1, g2, g3, g4-ext, cpol)
__device__ __forceinline__ void tdm_load_2d_bf16(u32 ldsByteOff, const void* gptr,
                                                 u32 rowElems, u32 rows,
                                                 u64 rowStrideElems) {
  const u64 ga = (u64)gptr;
  u32x4 g0;
  g0[0] = 1u;                                    // count=1, user descriptor
  g0[1] = ldsByteOff;                            // lds_addr (bytes)
  g0[2] = (u32)(ga & 0xffffffffu);               // global_addr[31:0]
  g0[3] = (u32)((ga >> 32) & 0x01ffffffu) | (2u << 30);  // addr[56:32] | type=2
  const u64 td0 = rowStrideElems;                // tensor_dim0 (no OOB in-row)
  const u32 td1 = rows;
  i32x8 g1;
  g1[0] = (int)(1u << 16);                                   // data_size=1 (2B)
  g1[1] = (int)(((u32)(td0 & 0xffffu)) << 16);               // dim0[15:0]
  g1[2] = (int)(((u32)((td0 >> 16) & 0xffffu)) | ((td1 & 0xffffu) << 16));
  g1[3] = (int)(((td1 >> 16) & 0xffffu) | ((rowElems & 0xffffu) << 16)); // tile_dim0
  g1[4] = (int)(rows & 0xffffu);                             // tile_dim1, tile_dim2=0
  g1[5] = (int)(rowStrideElems & 0xffffffffu);               // dim0_stride[31:0]
  g1[6] = (int)((rowStrideElems >> 32) & 0xffffu);           // dim0_stride[47:32]
  g1[7] = 0;
  i32x4 z4 = {0, 0, 0, 0};
  i32x8 z8 = {0, 0, 0, 0, 0, 0, 0, 0};
  __builtin_amdgcn_tensor_load_to_lds(g0, g1, z4, z4, z8, 0);
}
#endif

// ---------------------------------------------------------------------------
// One K=512 slab of the per-WG GEMM. A (64x512 bf16) lives in LDS; B rows are
// this wave's 16 gate columns read straight from the L2-resident packed
// weights (K-major, row stride KT).  acc[m] = 16x16 f32 tile (M-block m).
// All 4 A fragments of a k-step are staged into distinct locals BEFORE the
// WMMAs so the 8 ds_load_b128 issue back-to-back and drain with partial
// s_wait_dscnt counts, overlapping LDS latency with WMMA execution.
// ---------------------------------------------------------------------------
__device__ __forceinline__ void gemm_part(const bf16* Asrc, const bf16* Bsrc,
                                          v8f acc[4], int nb, int lane) {
  const int lrow = lane & 15;
  const int klo  = (lane >> 4) << 3;
  const bf16* brow  = Bsrc + (size_t)(nb * 16 + lrow) * KT + klo;
  const bf16* abase = Asrc + lrow * 512 + klo;
  for (int kk = 0; kk < 512; kk += 32) {
    v16bf b = frag_ld(brow + kk);
    v16bf a0 = frag_ld(abase + 0 * 16 * 512 + kk);
    v16bf a1 = frag_ld(abase + 1 * 16 * 512 + kk);
    v16bf a2 = frag_ld(abase + 2 * 16 * 512 + kk);
    v16bf a3 = frag_ld(abase + 3 * 16 * 512 + kk);
    acc[0] = __builtin_amdgcn_wmma_f32_16x16x32_bf16(false, a0, false, b, (short)0, acc[0], false, false);
    acc[1] = __builtin_amdgcn_wmma_f32_16x16x32_bf16(false, a1, false, b, (short)0, acc[1], false, false);
    acc[2] = __builtin_amdgcn_wmma_f32_16x16x32_bf16(false, a2, false, b, (short)0, acc[2], false, false);
    acc[3] = __builtin_amdgcn_wmma_f32_16x16x32_bf16(false, a3, false, b, (short)0, acc[3], false, false);
  }
}

// ---------------------------------------------------------------------------
// Grid-wide sync.  If dispatched as a workgroup cluster (ClusterID != 0) the
// whole 16-WG grid is one cluster: one wave signals the cluster barrier (-3),
// all waves wait on it.  Otherwise: sense barrier on an L2 atomic counter.
// ---------------------------------------------------------------------------
__device__ __forceinline__ void grid_sync(u32* bar, u32& phase, bool inCluster) {
  __threadfence();                 // release prior global h stores (DEV scope)
  __syncthreads();
  ++phase;
  if (inCluster) {
    if ((threadIdx.x >> 5) == 0)
      asm volatile("s_barrier_signal -3" ::: "memory");
    asm volatile("s_barrier_wait -3" ::: "memory");
  } else if (threadIdx.x == 0) {
    atomicAdd(bar, 1u);
    const u32 tgt = phase * NWG;
    while (__hip_atomic_load(bar, __ATOMIC_ACQUIRE, __HIP_MEMORY_SCOPE_AGENT) < tgt)
      __builtin_amdgcn_s_sleep(2);
  }
  __syncthreads();
  __threadfence();                 // acquire: invalidate stale WGP$ lines
}

// ---------------------------------------------------------------------------
// One LSTM layer-step for this WG.
//   xsrc : 64 rows x 512 bf16, row stride xRowStride (elements)
//   hprev: 64x512 bf16 (row stride 512) or nullptr => zero initial h
//   WtL  : packed weights for this layer (GG x KT, K-major, gate-permuted)
//   hout : this layer's h buffer (64x512); we write cols [wg*32, wg*32+32)
//   hcatOut: optional second sink (row stride KT) for the FC concat operand
// ---------------------------------------------------------------------------
__device__ __forceinline__ void layer_step(
    const bf16* xsrc, u64 xRowStride, const bf16* hprev,
    const bf16* WtL, const float* biasL,
    bf16* hout, bf16* hcatOut,
    float* cst, bool zeroC,
    bf16* Axs, bf16* Ahs, float* gatesS, int wg)
{
  const int tid = threadIdx.x, wave = tid >> 5, lane = tid & 31;

  // ---- stage activation tiles into LDS -----------------------------------
#if USE_TDM
  if (wave == 0) {
    tdm_load_2d_bf16(lds_off(Axs), xsrc, DD, BB, xRowStride);
    __builtin_amdgcn_s_wait_tensorcnt(0);
  } else if (wave == 1 && hprev) {
    tdm_load_2d_bf16(lds_off(Ahs), hprev, HH, BB, (u64)HH);
    __builtin_amdgcn_s_wait_tensorcnt(0);
  }
#else
  for (int e = tid; e < BB * DD / 8; e += 256) {
    int r = e >> 6, c8 = e & 63;
    *(v8bf*)(Axs + r * DD + c8 * 8) = *(const v8bf*)(xsrc + (u64)r * xRowStride + c8 * 8);
  }
  if (hprev)
    for (int e = tid; e < BB * HH / 8; e += 256) {
      int r = e >> 6, c8 = e & 63;
      *(v8bf*)(Ahs + r * HH + c8 * 8) = *(const v8bf*)(hprev + r * HH + c8 * 8);
    }
#endif
  // warm WGP$/L2 with this WG's 256KB weight slice (1KB per thread)
  __builtin_prefetch(WtL + (size_t)wg * 128 * KT + (size_t)tid * 512, 0, 1);
  __syncthreads();

  // ---- GEMM: gates(64x128) = [x|h](64x1024) @ Wt_slice^T ------------------
  const v8f vz = {0.f, 0.f, 0.f, 0.f, 0.f, 0.f, 0.f, 0.f};
  v8f acc[4] = {vz, vz, vz, vz};
  const bf16* Bsl = WtL + (size_t)wg * 128 * KT;
  gemm_part(Axs, Bsl, acc, wave, lane);             // K = 0..511  (x part)
  if (hprev) gemm_part(Ahs, Bsl + DD, acc, wave, lane);  // K = 512..1023 (h)

  // spill f32 accumulators to LDS per the ISA C/D layout (M=v+8*(lane/16))
  const int lrow = lane & 15, mhi = (lane >> 4) << 3;
#pragma unroll
  for (int m = 0; m < 4; ++m)
#pragma unroll
    for (int v = 0; v < 8; ++v)
      gatesS[(m * 16 + v + mhi) * 128 + wave * 16 + lrow] = acc[m][v];
  __syncthreads();

  // ---- fp32 cell update ---------------------------------------------------
  const float* bb = biasL + wg * 128;
  for (int e = tid; e < BB * NC; e += 256) {
    int r = e >> 5, c = e & 31;
    float iv = gatesS[r * 128 + c]      + bb[c];
    float fv = gatesS[r * 128 + 32 + c] + bb[32 + c];
    float gv = gatesS[r * 128 + 64 + c] + bb[64 + c];
    float ov = gatesS[r * 128 + 96 + c] + bb[96 + c];
    float c0 = zeroC ? 0.f : cst[e];
    float si = 1.f / (1.f + __expf(-iv));
    float sf = 1.f / (1.f + __expf(-fv));
    float so = 1.f / (1.f + __expf(-ov));
    float cn = c0 * sf + si * tanhf(gv);
    float hv = so * tanhf(cn);
    cst[e] = cn;
    hout[(size_t)r * HH + wg * NC + c] = (bf16)hv;
    if (hcatOut) hcatOut[(size_t)r * KT + wg * NC + c] = (bf16)hv;
  }
  __syncthreads();
}

// ---------------------------------------------------------------------------
// Persistent LSTM kernel: 16 WGs x 256 threads, 176KB dynamic LDS.
// ---------------------------------------------------------------------------
__global__ void __launch_bounds__(256)
k_lstm(const bf16* __restrict__ Xbf, const bf16* __restrict__ Wt,
       const float* __restrict__ bias,
       bf16* h0a, bf16* h0b, bf16* h1a, bf16* h1b, bf16* hcat, u32* bar)
{
  extern __shared__ char smem[];
  bf16*  Axs    = (bf16*)smem;               // 64x512 bf16 = 64KB
  bf16*  Ahs    = (bf16*)(smem + 65536);     // 64KB
  float* gatesS = (float*)(smem + 131072);   // 64x128 f32 = 32KB
  float* cst0   = (float*)(smem + 163840);   // 64x32 f32 = 8KB
  float* cst1   = (float*)(smem + 172032);   // 8KB   (total 180224B)
  const int wg = blockIdx.x;
  const bool inCluster = (__builtin_amdgcn_cluster_id_x() != 0);
  u32 phase = 0;

  for (int e = threadIdx.x; e < BB * NC; e += 256) { cst0[e] = 0.f; cst1[e] = 0.f; }
  __syncthreads();

  bf16* H0[2] = {h0a, h0b};
  bf16* H1[2] = {h1a, h1b};
  const bf16*  Wt0 = Wt;
  const bf16*  Wt1 = Wt + (size_t)GG * KT;
  const float* b0  = bias;
  const float* b1  = bias + GG;

  for (int t = 0; t < TT; ++t) {
    const int rd = t & 1, wr = rd ^ 1;
    layer_step(Xbf + (size_t)t * DD, (u64)TT * DD, H0[rd], Wt0, b0,
               H0[wr], nullptr, cst0, false, Axs, Ahs, gatesS, wg);
    grid_sync(bar, phase, inCluster);
    layer_step(H0[wr], (u64)HH, H1[rd], Wt1, b1,
               H1[wr], (t == TT - 1) ? hcat : nullptr, cst1, false,
               Axs, Ahs, gatesS, wg);
    grid_sync(bar, phase, inCluster);
  }

  // zero-state single step on x_{T-1} (reference's second branch)
  layer_step(Xbf + (size_t)(TT - 1) * DD, (u64)TT * DD, nullptr, Wt0, b0,
             H0[0], nullptr, cst0, true, Axs, Ahs, gatesS, wg);
  grid_sync(bar, phase, inCluster);
  layer_step(H0[0], (u64)HH, nullptr, Wt1, b1,
             H1[1], hcat + HH, cst1, true, Axs, Ahs, gatesS, wg);
}

// ---------------------------------------------------------------------------
// Final FC: out(64x512) = hcat(64x1024) @ Wfc + bfc, WMMA bf16.
// ---------------------------------------------------------------------------
__global__ void __launch_bounds__(128)
k_fc(const bf16* __restrict__ hcat, const bf16* __restrict__ WfcT,
     const float* __restrict__ bfc, float* __restrict__ out)
{
  const int g = blockIdx.x;             // cols [16g, 16g+16)
  const int wave = threadIdx.x >> 5;    // rows [16w, 16w+16)
  const int lane = threadIdx.x & 31;
  const int lrow = lane & 15, klo = (lane >> 4) << 3;
  v8f acc = {0.f, 0.f, 0.f, 0.f, 0.f, 0.f, 0.f, 0.f};
  const bf16* arow = hcat + (size_t)(wave * 16 + lrow) * KT + klo;
  const bf16* brow = WfcT + (size_t)(g * 16 + lrow) * KT + klo;
  for (int kk = 0; kk < KT; kk += 32) {
    v16bf a = frag_ld(arow + kk);
    v16bf b = frag_ld(brow + kk);
    acc = __builtin_amdgcn_wmma_f32_16x16x32_bf16(false, a, false, b, (short)0,
                                                  acc, false, false);
  }
  const int col = g * 16 + lrow;
  const float bv = bfc[col];
  const int mhi = (lane >> 4) << 3;
#pragma unroll
  for (int v = 0; v < 8; ++v)
    out[(size_t)(wave * 16 + v + mhi) * OO + col] = acc[v] + bv;
}

// ---------------------------------------------------------------------------
// Pack/convert kernels.
// ---------------------------------------------------------------------------
__global__ void __launch_bounds__(256)
k_cvt(const float* __restrict__ in, bf16* __restrict__ outp, int n) {
  for (int i = blockIdx.x * 256 + threadIdx.x; i < n; i += gridDim.x * 256)
    outp[i] = (bf16)in[i];
}

// Wt[l][n][k]: K-major rows; gate column n permuted so WG wg reads rows
// [wg*128, wg*128+128) = {i,f,g,o} x 32 contiguous h-columns.  bias fused.
__global__ void __launch_bounds__(256)
k_packw(const float* __restrict__ Wxh, const float* __restrict__ Whh,
        const float* __restrict__ bxh, const float* __restrict__ bhh,
        bf16* __restrict__ Wt, float* __restrict__ bias)
{
  const int total = 2 * GG * KT;
  for (int idx = blockIdx.x * 256 + threadIdx.x; idx < total; idx += gridDim.x * 256) {
    int l = idx / (GG * KT);
    int rem = idx % (GG * KT);
    int n = rem / KT, k = rem % KT;
    int wgp = n >> 7, r = n & 127, q = r >> 5, c = r & 31;
    int oc = q * HH + wgp * NC + c;                  // original gate column
    float v = (k < DD) ? Wxh[(size_t)l * DD * GG + (size_t)k * GG + oc]
                       : Whh[(size_t)l * HH * GG + (size_t)(k - DD) * GG + oc];
    Wt[idx] = (bf16)v;
  }
  for (int idx = blockIdx.x * 256 + threadIdx.x; idx < 2 * GG; idx += gridDim.x * 256) {
    int l = idx / GG, n = idx % GG;
    int wgp = n >> 7, r = n & 127, q = r >> 5, c = r & 31;
    int oc = q * HH + wgp * NC + c;
    bias[idx] = bxh[l * GG + oc] + bhh[l * GG + oc];
  }
}

__global__ void __launch_bounds__(256)
k_packfc(const float* __restrict__ Wfc, bf16* __restrict__ WfcT) {
  for (int i = blockIdx.x * 256 + threadIdx.x; i < OO * KT; i += gridDim.x * 256) {
    int o = i / KT, k = i % KT;
    WfcT[i] = (bf16)Wfc[(size_t)k * OO + o];
  }
}

__global__ void __launch_bounds__(256)
k_zero(u32* __restrict__ p, int n) {
  for (int i = blockIdx.x * 256 + threadIdx.x; i < n; i += gridDim.x * 256)
    p[i] = 0u;
}

// ---------------------------------------------------------------------------
// Workspace layout (bytes); ~73.4 MB total.
// ---------------------------------------------------------------------------
static const size_t XBF_OFF  = 0;                              // 64MB  bf16 input
static const size_t WT_OFF   = XBF_OFF  + (size_t)BB * TT * DD * 2;   // 8MB
static const size_t BIAS_OFF = WT_OFF   + (size_t)2 * GG * KT * 2;    // 16KB
static const size_t WFCT_OFF = BIAS_OFF + (size_t)2 * GG * 4;         // 1MB
static const size_t H0A_OFF  = WFCT_OFF + (size_t)OO * KT * 2;
static const size_t H0B_OFF  = H0A_OFF  + (size_t)BB * HH * 2;
static const size_t H1A_OFF  = H0B_OFF  + (size_t)BB * HH * 2;
static const size_t H1B_OFF  = H1A_OFF  + (size_t)BB * HH * 2;
static const size_t HCAT_OFF = H1B_OFF  + (size_t)BB * HH * 2;
static const size_t BAR_OFF  = HCAT_OFF + (size_t)BB * KT * 2;
static const size_t ZERO_BYTES = (BAR_OFF + 256) - H0A_OFF;
static const int    SMEM_BYTES = 180224;

extern "C" void kernel_launch(void* const* d_in, const int* in_sizes, int n_in,
                              void* d_out, int out_size, void* d_ws, size_t ws_size,
                              hipStream_t stream)
{
  const float* input = (const float*)d_in[0];
  const float* Wxh   = (const float*)d_in[1];
  const float* bxh   = (const float*)d_in[2];
  const float* Whh   = (const float*)d_in[3];
  const float* bhh   = (const float*)d_in[4];
  const float* Wfc   = (const float*)d_in[5];
  const float* bfc   = (const float*)d_in[6];

  char* ws = (char*)d_ws;
  bf16*  Xbf  = (bf16*)(ws + XBF_OFF);
  bf16*  Wt   = (bf16*)(ws + WT_OFF);
  float* bias = (float*)(ws + BIAS_OFF);
  bf16*  WfcT = (bf16*)(ws + WFCT_OFF);
  bf16*  h0a  = (bf16*)(ws + H0A_OFF);
  bf16*  h0b  = (bf16*)(ws + H0B_OFF);
  bf16*  h1a  = (bf16*)(ws + H1A_OFF);
  bf16*  h1b  = (bf16*)(ws + H1B_OFF);
  bf16*  hcat = (bf16*)(ws + HCAT_OFF);
  u32*   bar  = (u32*)(ws + BAR_OFF);

  (void)in_sizes; (void)n_in; (void)out_size; (void)ws_size;

  // allow >64KB dynamic LDS (gfx1250 WGP has 320KB)
  (void)hipFuncSetAttribute((const void*)k_lstm,
                            hipFuncAttributeMaxDynamicSharedMemorySize, SMEM_BYTES);

  k_cvt   <<<2048, 256, 0, stream>>>(input, Xbf, BB * TT * DD);
  k_packw <<<2048, 256, 0, stream>>>(Wxh, Whh, bxh, bhh, Wt, bias);
  k_packfc<<< 512, 256, 0, stream>>>(Wfc, WfcT);
  k_zero  <<< 128, 256, 0, stream>>>((u32*)(ws + H0A_OFF), (int)(ZERO_BYTES / 4));
  k_lstm  <<< NWG, 256, SMEM_BYTES, stream>>>(Xbf, Wt, bias, h0a, h0b, h1a, h1b, hcat, bar);
  k_fc    <<< OO / 16, 128, 0, stream>>>(hcat, WfcT, bfc, (float*)d_out);
}
